// VariationalRnn_68917045231647
// MI455X (gfx1250) — matrix operand order repaired and checked by reading, not compiled, not run on hardware
//
#include <hip/hip_runtime.h>
#include <hip/hip_bf16.h>
#include <math.h>

// ---------------------------------------------------------------------------
// MI455X (gfx1250) fused persistent VariationalRNN.
//   B=128, T=4096, H=128, K=5.  8 workgroups x 16 batch rows, 512 thr = 16 waves
//   (4 waves/SIMD -> full 256-VGPR budget per wave, no spills).
//   Waves 0-7 own N-tiles {w, 16+w}; waves 8-15 own tile {w}.  Wh_e / Wh_d held
//   as bf16 WMMA B-fragments in VGPRs; Wi_d' streamed per step from a
//   fragment-swizzled workspace (L2-resident); GRU state in LDS; all GEMMs via
//   v_wmma_f32_16x16x32_bf16.  Wave-uniform predicates go through
//   readfirstlane so WMMA regions are entered with EXEC all-ones (ISA 7.12).
//   Heads GEMM + coupling for step t-1 pipelined into P1/P2 of step t
//   -> 4 barriers per timestep.
// ---------------------------------------------------------------------------

typedef __bf16 bf16_t;
typedef bf16_t v16bf __attribute__((ext_vector_type(16)));
typedef bf16_t v8bf  __attribute__((ext_vector_type(8)));
typedef float  v8f   __attribute__((ext_vector_type(8)));

#define B_   128
#define T_   4096
#define H_   128
#define NTH  512   // threads per block
#define CH   128   // x/eps staging chunk (timesteps)
#define HXP  136   // padded bf16 row stride for state buffers
#define GP   392   // padded f32 row stride for gate buffers
#define Z8   {0.f,0.f,0.f,0.f,0.f,0.f,0.f,0.f}

// ---- fragment helpers (wave32; A/B use contiguous k order, same on both sides; C/D per CDNA5 ISA 7.12.2) ----------------

// A (16x32 bf16): lane l -> row M=l%16; lanes<16 hold K=0..15,
// lanes>=16 hold K=16..31.
__device__ inline v16bf load_A16(const bf16_t* buf, int lane, int kt) {
    int m   = lane & 15;
    int klo = (lane < 16) ? 0 : 16;
    const bf16_t* p = buf + m * HXP + kt * 32 + klo;
    v8bf lo = *(const v8bf*)p;        // K = klo .. klo+7
    v8bf hi = *(const v8bf*)(p + 8);  // K = klo+8 .. klo+15
    return __builtin_shufflevector(lo, hi, 0,1,2,3,4,5,6,7,8,9,10,11,12,13,14,15);
}

// B fragments pre-swizzled: frag(nt,kt), lane l = 16 contiguous bf16 (32B).
__device__ inline v16bf load_Bswz(const v16bf* base, int nt, int kt, int lane) {
    return base[(nt * 4 + kt) * 32 + lane];
}

// C/D (16x16 f32): element v -> row M = v + 8*(lane>=16), col N = lane%16.
__device__ inline void store_C(float* dst, int stride, int col0, int lane, v8f acc) {
    int c  = col0 + (lane & 15);
    int r0 = (lane < 16) ? 0 : 8;
#pragma unroll
    for (int v = 0; v < 8; ++v)
        dst[(r0 + v) * stride + c] = acc[v];
}

__device__ inline v8f wmma_bf16(v16bf a, v16bf b, v8f c) {
    return __builtin_amdgcn_wmma_f32_16x16x32_bf16(false, a, false, b, (short)0, c, false, false);
}

__device__ inline float sigmoidf_(float x) { return 1.0f / (1.0f + expf(-x)); }
__device__ inline float softplusf_(float x) {
    return fmaxf(x, 0.0f) + log1pf(expf(-fabsf(x)));
}

// ---------------------------------------------------------------------------
// Weight conversion: f32 -> bf16, swizzled into WMMA B-fragment order.
//   ws bf16 elems: [0) Wh_e swz | [49152) Wh_d swz | [98304) Wi_d' swz
//                  | [147456) W_head swz (2nt x 4kt x 32lane x 16)
// B layout (32x16): lane l -> col N=l%16; lanes<16 hold K=0..15, lanes>=16 K=16..31.
// ---------------------------------------------------------------------------
__global__ void vrnn_convert(const float* Whe, const float* Whd, const float* Wid,
                             const float* Wsh, const float* Wlsc, const float* Wpi,
                             const float* Wmu, const float* Wls, bf16_t* ws) {
    const int NM = 24 * 4 * 32 * 16;  // 49152 elems per big matrix
    int i = blockIdx.x * blockDim.x + threadIdx.x;
    if (i < 3 * NM) {
        int mat = i / NM, j = i % NM;
        int e    = j & 15;
        int lane = (j >> 4) & 31;
        int kt   = (j >> 9) & 3;
        int nt   = j >> 11;
        int col  = nt * 16 + (lane & 15);
        int k    = kt * 32 + ((lane < 16) ? 0 : 16) + e;
        float v;
        if      (mat == 0) v = Whe[k * 384 + col];
        else if (mat == 1) v = Whd[k * 384 + col];
        else               v = Wid[(k + 1) * 384 + col];  // skip pred row 0
        ws[i] = (bf16_t)v;
    } else if (i < 3 * NM + 4096) {
        int j    = i - 3 * NM;
        int e    = j & 15;
        int lane = (j >> 4) & 31;
        int kt   = (j >> 9) & 3;
        int nt   = j >> 11;                     // 0..1
        int c    = nt * 16 + (lane & 15);       // 0..31
        int k    = kt * 32 + ((lane < 16) ? 0 : 16) + e;  // 0..127
        float v = 0.0f;
        if      (c == 0) v = Wsh[k];
        else if (c == 1) v = Wlsc[k];
        else if (c < 7)  v = Wpi[k * 5 + (c - 2)];
        else if (c < 12) v = Wmu[k * 5 + (c - 7)];
        else if (c < 17) v = Wls[k * 5 + (c - 12)];
        ws[i] = (bf16_t)v;
    }
}

// ---------------------------------------------------------------------------
// Fused persistent encoder+decoder kernel. grid=8 blocks (16 batch rows each).
// ---------------------------------------------------------------------------
__global__ __launch_bounds__(NTH, 1)
void vrnn_fused(const float* __restrict__ x, const float* __restrict__ eps,
                const float* __restrict__ Wi_e, const float* __restrict__ bi_e,
                const float* __restrict__ bhn_e,
                const float* __restrict__ Wi_d, const float* __restrict__ bi_d,
                const float* __restrict__ bhn_d,
                const float* __restrict__ b_shift, const float* __restrict__ b_lsc,
                const float* __restrict__ b_pi, const float* __restrict__ b_mu,
                const float* __restrict__ b_ls,
                const v16bf* __restrict__ wWhe, const v16bf* __restrict__ wWhd,
                const v16bf* __restrict__ wWidp, const v16bf* __restrict__ wWhead,
                float* __restrict__ out) {
    __shared__ float  s_gh[16 * GP];      // encoder gh (P1->P2), then dec gi (P3->P4)
    __shared__ float  s_ghd[16 * GP];     // decoder gh (P1->P4)
    __shared__ bf16_t s_he[16 * HXP];     // encoder hidden state (bf16)
    __shared__ bf16_t s_hd[16 * HXP];     // decoder hidden state
    __shared__ v16bf  s_whfrag[256];      // head weights, fragment-swizzled (8KB)
    __shared__ float  s_ho[16 * 32];      // head GEMM output (for step t-1)
    __shared__ float  s_xbuf[16 * CH];    // staged x   (chunk of CH steps, 8KB)
    __shared__ float  s_ebuf[16 * CH];    // staged eps (chunk of CH steps, 8KB)
    __shared__ float  s_wie[384], s_bie[384], s_wid0[384], s_bid[384];
    __shared__ float  s_bhne[128], s_bhnd[128], s_bh[32];
    __shared__ float  s_pred[16], s_eprev[16];

    const int tid  = threadIdx.x;
    const int wave = tid >> 5;
    const int lane = tid & 31;
    const int b0   = blockIdx.x * 16;
    // wave-uniform predicates as scalars -> s_cbranch, never exec-masked WMMA
    const int uTwo   = __builtin_amdgcn_readfirstlane((tid >> 5) < 8 ? 1 : 0);
    const int uHeads = __builtin_amdgcn_readfirstlane((tid >> 5) >= 14 ? 1 : 0);

    // ---- one-time init ----
    for (int i = tid; i < 384; i += NTH) {
        s_wie[i]  = Wi_e[i];
        s_bie[i]  = bi_e[i];
        s_wid0[i] = Wi_d[i];   // row 0 of Wi_d: the pred input channel
        s_bid[i]  = bi_d[i];
    }
    for (int i = tid; i < 128; i += NTH) { s_bhne[i] = bhn_e[i]; s_bhnd[i] = bhn_d[i]; }
    if (tid < 32) {
        float bb = 0.0f;
        if      (tid == 0) bb = b_shift[0];
        else if (tid == 1) bb = b_lsc[0];
        else if (tid < 7)  bb = b_pi[tid - 2];
        else if (tid < 12) bb = b_mu[tid - 7];
        else if (tid < 17) bb = b_ls[tid - 12];
        s_bh[tid] = bb;
    }
    for (int i = tid; i < 16 * HXP; i += NTH) { s_he[i] = (bf16_t)0.0f; s_hd[i] = (bf16_t)0.0f; }
    if (tid < 256) s_whfrag[tid] = wWhead[tid];
    if (tid < 16)  { s_pred[tid] = 0.0f; s_eprev[tid] = 0.0f; }

    // Persistent VGPR-resident recurrent weights:
    //   waves 0-7: 2 tiles x 2 mats x 4 kt x 8 = 128 VGPRs; waves 8-15: 64 VGPRs.
    v16bf Be0[4], Bd0[4];
    v16bf Be1[4] = {}, Bd1[4] = {};
#pragma unroll
    for (int kt = 0; kt < 4; ++kt) {
        Be0[kt] = load_Bswz(wWhe, wave, kt, lane);
        Bd0[kt] = load_Bswz(wWhd, wave, kt, lane);
    }
    if (uTwo) {
#pragma unroll
        for (int kt = 0; kt < 4; ++kt) {
            Be1[kt] = load_Bswz(wWhe, 16 + wave, kt, lane);
            Bd1[kt] = load_Bswz(wWhd, 16 + wave, kt, lane);
        }
    }
    __syncthreads();

    // GEMM over this wave's tile(s): A from Abuf, B from register array (pair).
    auto gemm2 = [&](const bf16_t* Abuf, const v16bf* B0, const v16bf* B1, float* dst) {
        if (uTwo) {
            v8f a0 = Z8, a1 = Z8;
#pragma unroll
            for (int kt = 0; kt < 4; ++kt) {
                v16bf a = load_A16(Abuf, lane, kt);   // shared across both N-tiles
                a0 = wmma_bf16(a, B0[kt], a0);
                a1 = wmma_bf16(a, B1[kt], a1);
            }
            store_C(dst, GP, wave * 16, lane, a0);
            store_C(dst, GP, (16 + wave) * 16, lane, a1);
        } else {
            v8f a0 = Z8;
#pragma unroll
            for (int kt = 0; kt < 4; ++kt)
                a0 = wmma_bf16(load_A16(Abuf, lane, kt), B0[kt], a0);
            store_C(dst, GP, wave * 16, lane, a0);
        }
    };

    // coupling for timestep tt (reads s_ho, s_eprev; writes out + s_pred)
    auto do_coupling = [&](int tt) {
        const int m = tid;
        const float* ho = &s_ho[m * 32];
        float e     = s_eprev[m];
        float shift = ho[0] + s_bh[0];
        float lsc   = ho[1] + s_bh[1];
        float pi[5], mu[5], ls[5];
#pragma unroll
        for (int k = 0; k < 5; ++k) {
            pi[k] = ho[2 + k]  + s_bh[2 + k];
            mu[k] = ho[7 + k]  + s_bh[7 + k];
            ls[k] = ho[12 + k] + s_bh[12 + k];
        }
        float pmax = pi[0];
#pragma unroll
        for (int k = 1; k < 5; ++k) pmax = fmaxf(pmax, pi[k]);
        float psum = 0.0f;
#pragma unroll
        for (int k = 0; k < 5; ++k) psum += expf(pi[k] - pmax);
        float plse = logf(psum) + pmax;

        float ac[5], ap[5], amaxc = -1e30f, amaxp = -1e30f;
#pragma unroll
        for (int k = 0; k < 5; ++k) {
            float lpi  = pi[k] - plse;
            float z    = (e - mu[k]) * expf(-ls[k]);
            float lcdf = -softplusf_(-z);                                  // log_sigmoid(z)
            float lpdf = -fabsf(z) - 2.0f * log1pf(expf(-fabsf(z))) - ls[k];
            ac[k] = lpi + lcdf;  ap[k] = lpi + lpdf;
            amaxc = fmaxf(amaxc, ac[k]);  amaxp = fmaxf(amaxp, ap[k]);
        }
        float sc_ = 0.0f, sp_ = 0.0f;
#pragma unroll
        for (int k = 0; k < 5; ++k) { sc_ += expf(ac[k] - amaxc); sp_ += expf(ap[k] - amaxp); }
        float log_cdf = logf(sc_) + amaxc;
        float log_pdf = logf(sp_) + amaxp;

        log_cdf = fminf(log_cdf, -1e-12f);
        float log_1m  = logf(-expm1f(log_cdf));
        float val     = log_cdf - log_1m;
        float log_jac = log_pdf - log_cdf - log_1m;

        float prediction = shift + val * expf(lsc);
        float logp = (-e - 2.0f * softplusf_(-e)) - lsc - log_jac;

        out[(b0 + m) * T_ + tt]           = logp;        // logp.T  [B,T]
        out[B_ * T_ + (b0 + m) * T_ + tt] = prediction;  // preds.T [B,T]
        s_pred[m] = prediction;
    };

    // heads GEMM on current s_hd -> s_ho (waves 14,15; single-tile waves)
    auto do_heads = [&]() {
        const int nt = wave - 14;
        v8f acc = Z8;
#pragma unroll
        for (int kt = 0; kt < 4; ++kt)
            acc = wmma_bf16(load_A16(s_hd, lane, kt),
                            s_whfrag[(nt * 4 + kt) * 32 + lane], acc);
        store_C(s_ho, 32, nt * 16, lane, acc);
    };

    for (int t = 0; t < T_; ++t) {
        // ---- P1: recurrent GEMMs (prev-step states) + heads(t-1) + chunk staging ----
        gemm2(s_he, Be0, Be1, s_gh);    // encoder  gh = h_e(t-1) @ Wh_e
        gemm2(s_hd, Bd0, Bd1, s_ghd);   // decoder  gh = h_d(t-1) @ Wh_d
        if (uHeads) do_heads();         // heads(h_d(t-1)); garbage-but-unused at t==0
        if ((t & (CH - 1)) == 0) {      // stage next CH steps of x / eps
            const int c0 = t;
            for (int i = tid; i < 16 * CH; i += NTH) {
                int m = i >> 7, j = i & (CH - 1);
                s_xbuf[i] = x[(b0 + m) * T_ + (T_ - 1 - (c0 + j))];
                s_ebuf[i] = eps[(b0 + m) * T_ + (c0 + j)];
            }
        }
        __syncthreads();

        // ---- P2: encoder gates (waves 1..15)  ||  coupling for step t-1 (lanes 0..15) ----
        if (tid >= 32) {
            for (int i = tid - 32; i < 2048; i += (NTH - 32)) {
                int m = i >> 7, n = i & 127;
                float xv  = s_xbuf[m * CH + (t & (CH - 1))];
                float gr  = fmaf(xv, s_wie[n],       s_bie[n])       + s_gh[m * GP + n];
                float gz  = fmaf(xv, s_wie[128 + n], s_bie[128 + n]) + s_gh[m * GP + 128 + n];
                float gn  = fmaf(xv, s_wie[256 + n], s_bie[256 + n]);
                float ghn = s_gh[m * GP + 256 + n] + s_bhne[n];
                float r   = sigmoidf_(gr);
                float zg  = sigmoidf_(gz);
                float nn  = tanhf(gn + r * ghn);
                float hp  = (float)s_he[m * HXP + n];
                s_he[m * HXP + n] = (bf16_t)((1.0f - zg) * nn + zg * hp);
            }
        } else if (tid < 16 && t > 0) {
            do_coupling(t - 1);
        }
        __syncthreads();

        // ---- P3: decoder input GEMM  gi = h_enc @ Wi_d'  (B streamed from L2) ----
        if (uTwo) {
            v8f a0 = Z8, a1 = Z8;
#pragma unroll
            for (int kt = 0; kt < 4; ++kt) {
                v16bf a = load_A16(s_he, lane, kt);
                a0 = wmma_bf16(a, load_Bswz(wWidp, wave, kt, lane), a0);
                a1 = wmma_bf16(a, load_Bswz(wWidp, 16 + wave, kt, lane), a1);
            }
            store_C(s_gh, GP, wave * 16, lane, a0);
            store_C(s_gh, GP, (16 + wave) * 16, lane, a1);
        } else {
            v8f a0 = Z8;
#pragma unroll
            for (int kt = 0; kt < 4; ++kt)
                a0 = wmma_bf16(load_A16(s_he, lane, kt),
                               load_Bswz(wWidp, wave, kt, lane), a0);
            store_C(s_gh, GP, wave * 16, lane, a0);
        }
        __syncthreads();

        // ---- P4: decoder gates (+ pred(t-1) feedback through Wi_d row 0) ----
        for (int i = tid; i < 2048; i += NTH) {
            int m = i >> 7, n = i & 127;
            float p   = s_pred[m];
            float gr  = fmaf(p, s_wid0[n],       s_bid[n])       + s_gh[m * GP + n]       + s_ghd[m * GP + n];
            float gz  = fmaf(p, s_wid0[128 + n], s_bid[128 + n]) + s_gh[m * GP + 128 + n] + s_ghd[m * GP + 128 + n];
            float gn  = fmaf(p, s_wid0[256 + n], s_bid[256 + n]) + s_gh[m * GP + 256 + n];
            float ghn = s_ghd[m * GP + 256 + n] + s_bhnd[n];
            float r   = sigmoidf_(gr);
            float zg  = sigmoidf_(gz);
            float nn  = tanhf(gn + r * ghn);
            float hp  = (float)s_hd[m * HXP + n];
            s_hd[m * HXP + n] = (bf16_t)((1.0f - zg) * nn + zg * hp);
        }
        if (tid < 16)  // snapshot eps(t) before the staging buffer can roll over
            s_eprev[tid] = s_ebuf[tid * CH + (t & (CH - 1))];
        __syncthreads();
    }

    // ---- epilogue: heads + coupling for the final step t = T-1 ----
    if (uHeads) do_heads();
    __syncthreads();
    if (tid < 16) do_coupling(T_ - 1);
}

// ---------------------------------------------------------------------------
extern "C" void kernel_launch(void* const* d_in, const int* in_sizes, int n_in,
                              void* d_out, int out_size, void* d_ws, size_t ws_size,
                              hipStream_t stream) {
    // setup_inputs() order:
    // 0:s 1:x 2:eps 3:Wi_e 4:Wh_e 5:bi_e 6:bhn_e 7:Wi_d 8:Wh_d 9:bi_d 10:bhn_d
    // 11:W_shift 12:b_shift 13:W_lsc 14:b_lsc 15:W_pi 16:b_pi 17:W_mu 18:b_mu 19:W_ls 20:b_ls
    const float* x     = (const float*)d_in[1];
    const float* eps   = (const float*)d_in[2];
    const float* Wi_e  = (const float*)d_in[3];
    const float* Wh_e  = (const float*)d_in[4];
    const float* bi_e  = (const float*)d_in[5];
    const float* bhn_e = (const float*)d_in[6];
    const float* Wi_d  = (const float*)d_in[7];
    const float* Wh_d  = (const float*)d_in[8];
    const float* bi_d  = (const float*)d_in[9];
    const float* bhn_d = (const float*)d_in[10];
    const float* Wsh   = (const float*)d_in[11];
    const float* bsh   = (const float*)d_in[12];
    const float* Wlsc  = (const float*)d_in[13];
    const float* blsc  = (const float*)d_in[14];
    const float* Wpi   = (const float*)d_in[15];
    const float* bpi   = (const float*)d_in[16];
    const float* Wmu   = (const float*)d_in[17];
    const float* bmu   = (const float*)d_in[18];
    const float* Wls   = (const float*)d_in[19];
    const float* bls   = (const float*)d_in[20];

    bf16_t* ws = (bf16_t*)d_ws;
    const v16bf* wWhe   = (const v16bf*)(ws);
    const v16bf* wWhd   = (const v16bf*)(ws + 49152);
    const v16bf* wWidp  = (const v16bf*)(ws + 2 * 49152);
    const v16bf* wWhead = (const v16bf*)(ws + 3 * 49152);

    const int nconv = 3 * 49152 + 4096;
    vrnn_convert<<<(nconv + 255) / 256, 256, 0, stream>>>(
        Wh_e, Wh_d, Wi_d, Wsh, Wlsc, Wpi, Wmu, Wls, ws);

    vrnn_fused<<<B_ / 16, NTH, 0, stream>>>(
        x, eps, Wi_e, bi_e, bhn_e, Wi_d, bi_d, bhn_d,
        bsh, blsc, bpi, bmu, bls,
        wWhe, wWhd, wWidp, wWhead, (float*)d_out);
}
